// QuantizedDenseNet_4114578669854
// MI455X (gfx1250) — matrix operand order repaired and checked by reading, not compiled, not run on hardware
//
#include <hip/hip_runtime.h>
#include <hip/hip_bf16.h>

// ---------------------------------------------------------------------------
// Types for CDNA5 WMMA
// ---------------------------------------------------------------------------
typedef __attribute__((ext_vector_type(16))) _Float16 v16h;
typedef __attribute__((ext_vector_type(8)))  _Float16 v8h;
typedef __attribute__((ext_vector_type(8)))  float    v8f;

#define QMAX   255.0f
#define ACT_S  0.05f
#define IN_S   (4.0f / 255.0f)
#define IN_ZP  128.0f
#define EPS_BN 1e-5f

#define CONV_BM 64
#define CONV_KC 32
#define ROWH    (CONV_KC + 8)   // LDS row stride in halves (80 B, 16B-aligned)

__device__ __forceinline__ v8h zero8h() {
  v8h z;
#pragma unroll
  for (int i = 0; i < 8; ++i) z[i] = (_Float16)0.f;
  return z;
}

__device__ __forceinline__ int clampi(int v, int hi) { return v > hi ? hi : v; }

// ---------------------------------------------------------------------------
// Shared WMMA tile helpers (A in Asm[m][k], B transposed in BsmT[n][k];
// every fragment is two aligned ds_load_b128, zero repacking).
// ---------------------------------------------------------------------------
template <int NT>
__device__ __forceinline__ void wmma_tile(const _Float16 (*Asm)[ROWH],
                                          const _Float16 (*BsmT)[ROWH],
                                          v8f acc[NT], int arow, int l15, int kh) {
  union { v16h v; v8h h[2]; } af;
  af.h[0] = *(const v8h*)&Asm[arow][kh * 8];
  af.h[1] = *(const v8h*)&Asm[arow][16 + kh * 8];
#pragma unroll
  for (int nt = 0; nt < NT; ++nt) {
    union { v16h v; v8h h[2]; } bf;
    const _Float16* bp = &BsmT[nt * 16 + l15][kh * 8];
    bf.h[0] = *(const v8h*)bp;
    bf.h[1] = *(const v8h*)(bp + 16);
    acc[nt] = __builtin_amdgcn_wmma_f32_16x16x32_f16(
        false, af.v, false, bf.v, (short)0, acc[nt], false, false);
  }
}

template <int NT>
__device__ __forceinline__ void store_tile(const v8f acc[NT],
                                           _Float16* __restrict__ out,
                                           int m0, int wave, int kh, int l15,
                                           int n0, int Mtot, int Cout,
                                           int CS, int cofs) {
#pragma unroll
  for (int nt = 0; nt < NT; ++nt) {
#pragma unroll
    for (int r = 0; r < 8; ++r) {
      int m = m0 + wave * 16 + r + 8 * kh;
      int n = n0 + nt * 16 + l15;
      if (m < Mtot && n < Cout)
        out[(size_t)m * CS + cofs + n] = (_Float16)acc[nt][r];
    }
  }
}

// ---------------------------------------------------------------------------
// 1x1 convolution over NHWC = plain GEMM.  Staging is fully unconditional:
// out-of-range rows/cols are CLAMPED (their garbage results are masked at the
// store), K = Cin is a multiple of 32, weights are K-padded.  Double-buffered
// LDS: one barrier per K chunk, global->LDS of chunk i+1 overlaps WMMA of i.
// ---------------------------------------------------------------------------
__global__ __launch_bounds__(128)
void conv1x1_nhwc(const _Float16* __restrict__ act,  // [Mtot][Cin]
                  const _Float16* __restrict__ wq,   // [Cout][Kpad]
                  _Float16* __restrict__ out,        // [Mtot][CS] @ cofs
                  int Cin, int Mtot, int Cout, int CS, int cofs, int Kpad) {
  __shared__ __align__(16) _Float16 Asm[2][CONV_BM][ROWH];
  __shared__ __align__(16) _Float16 BsmT[2][64][ROWH];

  const int tid  = threadIdx.x;
  const int lane = tid & 31, wave = tid >> 5;
  const int l15  = lane & 15, kh = lane >> 4;
  const int m0 = blockIdx.x * CONV_BM;
  const int n0 = blockIdx.y * 64;

  v8f acc[4];
#pragma unroll
  for (int nt = 0; nt < 4; ++nt)
    acc[nt] = (v8f){0.f, 0.f, 0.f, 0.f, 0.f, 0.f, 0.f, 0.f};

  const int arow_ld = tid >> 1;
  const int akb     = (tid & 1) * 16;
  const _Float16* asrc =
      act + (size_t)clampi(m0 + arow_ld, Mtot - 1) * Cin + akb;

  const int bn_ld = tid >> 1;
  const _Float16* bsrc =
      wq + (size_t)clampi(n0 + bn_ld, Cout - 1) * Kpad + akb;

  const int arow = wave * 16 + l15;

  // stage chunk 0
  *(v8h*)&Asm[0][arow_ld][akb]      = *(const v8h*)(asrc);
  *(v8h*)&Asm[0][arow_ld][akb + 8]  = *(const v8h*)(asrc + 8);
  *(v8h*)&BsmT[0][bn_ld][akb]       = *(const v8h*)(bsrc);
  *(v8h*)&BsmT[0][bn_ld][akb + 8]   = *(const v8h*)(bsrc + 8);
  __syncthreads();

  const int nk = Cin / CONV_KC;
  for (int i = 0; i < nk; ++i) {
    const int cur = i & 1;
    if (i + 1 < nk) {
      const int k1 = (i + 1) * CONV_KC;
      *(v8h*)&Asm[cur ^ 1][arow_ld][akb]     = *(const v8h*)(asrc + k1);
      *(v8h*)&Asm[cur ^ 1][arow_ld][akb + 8] = *(const v8h*)(asrc + k1 + 8);
      *(v8h*)&BsmT[cur ^ 1][bn_ld][akb]      = *(const v8h*)(bsrc + k1);
      *(v8h*)&BsmT[cur ^ 1][bn_ld][akb + 8]  = *(const v8h*)(bsrc + k1 + 8);
      __builtin_prefetch((const void*)(bsrc + k1 + CONV_KC), 0, 1);
    }
    wmma_tile<4>(Asm[cur], BsmT[cur], acc, arow, l15, kh);
    __syncthreads();
  }
  store_tile<4>(acc, out, m0, wave, kh, l15, n0, Mtot, Cout, CS, cofs);
}

// ---------------------------------------------------------------------------
// 3x3 s1 p1 convolution over NHWC with Cin = 128.  K-order (r,s,c): each
// 32-wide K chunk lies inside one filter tap -> single whole-span spatial
// guard (required for zero padding); M/N are clamped.  Double-buffered LDS.
// ---------------------------------------------------------------------------
__global__ __launch_bounds__(128)
void conv3x3_c128_nhwc(const _Float16* __restrict__ act, // [16][H][W][128]
                       const _Float16* __restrict__ wq,  // [Cout][1152 pad]
                       _Float16* __restrict__ out,       // [Mtot][CS] @ cofs
                       int H, int W, int Cout, int CS, int cofs, int Kpad) {
  __shared__ __align__(16) _Float16 Asm[2][CONV_BM][ROWH];
  __shared__ __align__(16) _Float16 BsmT[2][32][ROWH];

  const int HW   = H * W;
  const int Mtot = 16 * HW;
  const int Ktot = 9 * 128;

  const int tid  = threadIdx.x;
  const int lane = tid & 31, wave = tid >> 5;
  const int l15  = lane & 15, kh = lane >> 4;
  const int m0 = blockIdx.x * CONV_BM;
  const int n0 = blockIdx.y * 32;

  v8f acc[2];
#pragma unroll
  for (int nt = 0; nt < 2; ++nt)
    acc[nt] = (v8f){0.f, 0.f, 0.f, 0.f, 0.f, 0.f, 0.f, 0.f};

  const int arow_ld = tid >> 1;
  const int akb     = (tid & 1) * 16;
  const int am      = clampi(m0 + arow_ld, Mtot - 1);
  const int abb = am / HW;
  const int ap  = am - abb * HW;
  const int aoy = ap / W;
  const int aox = ap - aoy * W;

  const int bn_ld = tid >> 2;
  const int bkb   = (tid & 3) * 8;
  const _Float16* bsrc =
      wq + (size_t)clampi(n0 + bn_ld, Cout - 1) * Kpad + bkb;

  const int arow = wave * 16 + l15;

  auto stageA = [&](int k0, int buf) {
    const int rs = k0 >> 7;                  // tap 0..8 (Cin = 128)
    const int c0 = k0 & 127;
    const int r  = rs / 3, s = rs - 3 * r;   // const divisions
    const int iy = aoy + r - 1, ix = aox + s - 1;
    v8h alo = zero8h(), ahi = zero8h();
    if ((unsigned)iy < (unsigned)H && (unsigned)ix < (unsigned)W) {
      const _Float16* apn =
          act + (((size_t)(abb * H + iy) * W + ix) << 7) + c0 + akb;
      alo = *(const v8h*)apn;
      ahi = *(const v8h*)(apn + 8);
    }
    *(v8h*)&Asm[buf][arow_ld][akb]     = alo;
    *(v8h*)&Asm[buf][arow_ld][akb + 8] = ahi;
  };

  stageA(0, 0);
  *(v8h*)&BsmT[0][bn_ld][bkb] = *(const v8h*)(bsrc);
  __syncthreads();

  const int nk = Ktot / CONV_KC;
  for (int i = 0; i < nk; ++i) {
    const int cur = i & 1;
    if (i + 1 < nk) {
      const int k1 = (i + 1) * CONV_KC;
      stageA(k1, cur ^ 1);
      *(v8h*)&BsmT[cur ^ 1][bn_ld][bkb] = *(const v8h*)(bsrc + k1);
      __builtin_prefetch((const void*)(bsrc + k1 + CONV_KC), 0, 1);
    }
    wmma_tile<2>(Asm[cur], BsmT[cur], acc, arow, l15, kh);
    __syncthreads();
  }
  store_tile<2>(acc, out, m0, wave, kh, l15, n0, Mtot, Cout, CS, cofs);
}

// ---------------------------------------------------------------------------
// Stem 7x7 s2 p3 conv over NHWC (Cin=3).  One-off kernel: per-element gather
// with constant-divisor K decomposition (K-order (r,s,c), Ktot=147, 5 chunks).
// M is a multiple of 64 here (16*112*112), so only K/spatial guards remain.
// ---------------------------------------------------------------------------
__global__ __launch_bounds__(128)
void conv7x7s2_nhwc(const _Float16* __restrict__ act, // [16][224][224][3]
                    const _Float16* __restrict__ wq,  // [64][Kpad=160]
                    _Float16* __restrict__ out,       // [Mtot][64]
                    int Kpad) {
  __shared__ __align__(16) _Float16 Asm[CONV_BM][ROWH];
  __shared__ __align__(16) _Float16 BsmT[64][ROWH];

  const int Ho = 112, Wo = 112, H = 224, W = 224;
  const int HWo = Ho * Wo;
  const int Mtot = 16 * HWo;
  const int Ktot = 147;

  const int tid  = threadIdx.x;
  const int lane = tid & 31, wave = tid >> 5;
  const int l15  = lane & 15, kh = lane >> 4;
  const int m0 = blockIdx.x * CONV_BM;

  v8f acc[4];
#pragma unroll
  for (int nt = 0; nt < 4; ++nt)
    acc[nt] = (v8f){0.f, 0.f, 0.f, 0.f, 0.f, 0.f, 0.f, 0.f};

  const int arow_ld = tid >> 1;
  const int akb     = (tid & 1) * 16;
  const int am      = m0 + arow_ld;
  const int abb = am / HWo;
  const int ap  = am - abb * HWo;
  const int aoy = ap / Wo;
  const int aox = ap - aoy * Wo;

  const int bng = tid >> 1;            // Cout == 64, grid.y == 1
  const _Float16* bsrc = wq + (size_t)bng * Kpad + akb;

  const int arow = wave * 16 + l15;

  for (int k0 = 0; k0 < Ktot; k0 += CONV_KC) {
    v8h alo = zero8h(), ahi = zero8h();
#pragma unroll
    for (int i = 0; i < 16; ++i) {
      int k = k0 + akb + i;
      _Float16 v = (_Float16)0.f;
      if (k < Ktot) {
        int rs = k / 3;                // const divisions
        int c  = k - 3 * rs;
        int r  = rs / 7;
        int s  = rs - 7 * r;
        int iy = aoy * 2 - 3 + r;
        int ix = aox * 2 - 3 + s;
        if ((unsigned)iy < (unsigned)H && (unsigned)ix < (unsigned)W)
          v = act[((size_t)(abb * H + iy) * W + ix) * 3 + c];
      }
      if (i < 8) alo[i] = v; else ahi[i - 8] = v;
    }
    *(v8h*)&Asm[arow_ld][akb]     = alo;
    *(v8h*)&Asm[arow_ld][akb + 8] = ahi;
    *(v8h*)&BsmT[bng][akb]     = *(const v8h*)(bsrc + k0);
    *(v8h*)&BsmT[bng][akb + 8] = *(const v8h*)(bsrc + k0 + 8);
    __syncthreads();
    wmma_tile<4>(Asm, BsmT, acc, arow, l15, kh);
    __syncthreads();
  }
  store_tile<4>(acc, out, m0, wave, kh, l15, 0, Mtot, 64, 64, 0);
}

// ---------------------------------------------------------------------------
// Input fake-quant: NCHW fp32 -> NHWC fp16
// ---------------------------------------------------------------------------
__global__ void quant_input_kernel(const float* __restrict__ x,
                                   _Float16* __restrict__ out, int HW) {
  int i = blockIdx.x * blockDim.x + threadIdx.x;    // over 16*HW*3 (NHWC)
  if (i >= 16 * HW * 3) return;
  int c  = i % 3;
  int px = i / 3;
  int hw = px % HW;
  int b  = px / HW;
  float q = rintf(x[((size_t)(b * 3 + c) * HW) + hw] * (1.0f / IN_S)) + IN_ZP;
  q = fminf(fmaxf(q, 0.0f), QMAX);
  out[i] = (_Float16)((q - IN_ZP) * IN_S);
}

// ---------------------------------------------------------------------------
// Weight fake-quant + layout: OIHW fp32 -> [O][(r,s,c)] fp16, K padded to 32.
// ---------------------------------------------------------------------------
__global__ void zero_u32_kernel(unsigned* p) { *p = 0u; }

__global__ void maxabs_kernel(const float* __restrict__ w, int n,
                              unsigned* __restrict__ res) {
  __shared__ float sm[256];
  float m = 0.0f;
  for (int i = blockIdx.x * blockDim.x + threadIdx.x; i < n;
       i += gridDim.x * blockDim.x)
    m = fmaxf(m, fabsf(w[i]));
  sm[threadIdx.x] = m;
  __syncthreads();
  for (int s = 128; s > 0; s >>= 1) {
    if ((int)threadIdx.x < s)
      sm[threadIdx.x] = fmaxf(sm[threadIdx.x], sm[threadIdx.x + s]);
    __syncthreads();
  }
  if (threadIdx.x == 0) atomicMax(res, __float_as_uint(sm[0]));
}

__global__ void quant_weight_kernel(const float* __restrict__ w,
                                    const unsigned* __restrict__ mx,
                                    _Float16* __restrict__ out,
                                    int Cout, int C, int KH, int KW, int Kpad) {
  int i = blockIdx.x * blockDim.x + threadIdx.x;
  int Ktot = C * KH * KW;
  if (i >= Cout * Kpad) return;
  int n = i / Kpad;
  int k = i - n * Kpad;
  _Float16 o = (_Float16)0.f;
  if (k < Ktot) {
    int rs = k / C;                    // k = (r*KW+s)*C + c
    int c  = k - rs * C;
    int r  = rs / KW;
    int s  = rs - r * KW;
    float sc = fmaxf(__uint_as_float(*mx), 1e-8f) * 2.0f / QMAX;
    float q = rintf(w[((size_t)(n * C + c) * KH + r) * KW + s] / sc)
              + (QMAX * 0.5f);
    q = fminf(fmaxf(q, 0.0f), QMAX);
    o = (_Float16)((q - QMAX * 0.5f) * sc);
  }
  out[i] = o;
}

// ---------------------------------------------------------------------------
// Fused BN+ReLU+requant over NHWC, channel-vectorized (v8h in/out).
// ---------------------------------------------------------------------------
__global__ void bn_relu_q_kernel(const _Float16* __restrict__ in,
                                 const float* __restrict__ gamma,
                                 const float* __restrict__ beta,
                                 const float* __restrict__ mean,
                                 const float* __restrict__ var,
                                 _Float16* __restrict__ out,
                                 int C, int NPX, int inCS) {
  int i = blockIdx.x * blockDim.x + threadIdx.x;    // over NPX * (C/8)
  int cg = C >> 3;
  if (i >= NPX * cg) return;
  int px = i / cg;
  int c0 = (i - px * cg) << 3;
  v8h x = *(const v8h*)(in + (size_t)px * inCS + c0);
  v8h o;
#pragma unroll
  for (int j = 0; j < 8; ++j) {
    int c = c0 + j;
    float inv = gamma[c] * rsqrtf(var[c] + EPS_BN);
    float y = (float)x[j] * inv + (beta[c] - mean[c] * inv);
    y = fmaxf(y, 0.0f);
    float q = fminf(fmaxf(rintf(y * (1.0f / ACT_S)), 0.0f), QMAX);
    o[j] = (_Float16)(q * ACT_S);
  }
  *(v8h*)(out + (size_t)px * C + c0) = o;
}

// ---------------------------------------------------------------------------
// MaxPool 3x3 s2 p1 over NHWC, channel-vectorized. in compact C; out @ CS.
// ---------------------------------------------------------------------------
__global__ void maxpool3s2p1_kernel(const _Float16* __restrict__ in,
                                    _Float16* __restrict__ out,
                                    int C, int H, int W, int Ho, int Wo, int CS) {
  int cg = C >> 3;
  int i = blockIdx.x * blockDim.x + threadIdx.x;    // over 16*Ho*Wo*cg
  if (i >= 16 * Ho * Wo * cg) return;
  int c0 = (i % cg) << 3;
  int px = i / cg;
  int ox = px % Wo;
  int oy = (px / Wo) % Ho;
  int b  = px / (Wo * Ho);
  float m[8];
#pragma unroll
  for (int j = 0; j < 8; ++j) m[j] = -3.4e38f;
  for (int ky = 0; ky < 3; ++ky) {
    int iy = oy * 2 - 1 + ky;
    if ((unsigned)iy >= (unsigned)H) continue;
    for (int kx = 0; kx < 3; ++kx) {
      int ix = ox * 2 - 1 + kx;
      if ((unsigned)ix >= (unsigned)W) continue;
      v8h v = *(const v8h*)(in + ((size_t)(b * H + iy) * W + ix) * C + c0);
#pragma unroll
      for (int j = 0; j < 8; ++j) m[j] = fmaxf(m[j], (float)v[j]);
    }
  }
  v8h o;
#pragma unroll
  for (int j = 0; j < 8; ++j) o[j] = (_Float16)m[j];
  *(v8h*)(out + (size_t)px * CS + c0) = o;
}

// ---------------------------------------------------------------------------
// AvgPool 2x2 s2 over NHWC, channel-vectorized. in compact C; out @ CS.
// ---------------------------------------------------------------------------
__global__ void avgpool2_kernel(const _Float16* __restrict__ in,
                                _Float16* __restrict__ out,
                                int C, int H, int W, int CS) {
  int Ho = H >> 1, Wo = W >> 1;
  int cg = C >> 3;
  int i = blockIdx.x * blockDim.x + threadIdx.x;
  if (i >= 16 * Ho * Wo * cg) return;
  int c0 = (i % cg) << 3;
  int px = i / cg;
  int ox = px % Wo;
  int oy = (px / Wo) % Ho;
  int b  = px / (Wo * Ho);
  const _Float16* p00 = in + ((size_t)(b * H + oy * 2) * W + ox * 2) * C + c0;
  v8h a = *(const v8h*)p00;
  v8h bv = *(const v8h*)(p00 + C);
  v8h cv = *(const v8h*)(p00 + (size_t)W * C);
  v8h dv = *(const v8h*)(p00 + (size_t)W * C + C);
  v8h o;
#pragma unroll
  for (int j = 0; j < 8; ++j)
    o[j] = (_Float16)(((float)a[j] + (float)bv[j] + (float)cv[j] + (float)dv[j]) * 0.25f);
  *(v8h*)(out + (size_t)px * CS + c0) = o;
}

// ---------------------------------------------------------------------------
// Global average pool over NHWC: [16][HW][C] fp16 -> [16][C] fp32
// ---------------------------------------------------------------------------
__global__ void gpool_kernel(const _Float16* __restrict__ in,
                             float* __restrict__ out, int C, int HW) {
  int cg = C >> 3;
  int i = blockIdx.x * blockDim.x + threadIdx.x;    // over 16*cg
  if (i >= 16 * cg) return;
  int c0 = (i % cg) << 3;
  int b  = i / cg;
  float s[8];
#pragma unroll
  for (int j = 0; j < 8; ++j) s[j] = 0.0f;
  const _Float16* p = in + (size_t)b * HW * C + c0;
  for (int hw = 0; hw < HW; ++hw) {
    v8h v = *(const v8h*)(p + (size_t)hw * C);
#pragma unroll
    for (int j = 0; j < 8; ++j) s[j] += (float)v[j];
  }
#pragma unroll
  for (int j = 0; j < 8; ++j) out[(size_t)b * C + c0 + j] = s[j] / (float)HW;
}

// ---------------------------------------------------------------------------
// FC: out[16][N] = pooled[16][K] @ wq[N][K]^T + bias  (WMMA over K)
// ---------------------------------------------------------------------------
__global__ __launch_bounds__(32)
void fc_wmma_kernel(const float* __restrict__ pooled,   // [16][K]
                    const _Float16* __restrict__ wq,    // [N][K], K%32==0
                    const float* __restrict__ bias,
                    float* __restrict__ out, int K, int N) {
  const int lane = threadIdx.x;
  const int l15  = lane & 15;
  const int kh   = lane >> 4;
  const int n0   = blockIdx.x * 16;
  const int n    = n0 + l15;
  const int nc   = n > N - 1 ? N - 1 : n;   // clamp, mask at store
  v8f acc = (v8f){0.f, 0.f, 0.f, 0.f, 0.f, 0.f, 0.f, 0.f};
  for (int k0 = 0; k0 < K; k0 += 32) {
    union { v16h v; v8h h[2]; } af, bf;
#pragma unroll
    for (int i = 0; i < 8; ++i) {
      af.v[i]     = (_Float16)pooled[l15 * K + k0 + kh * 8 + i];
      af.v[8 + i] = (_Float16)pooled[l15 * K + k0 + 16 + kh * 8 + i];
    }
    bf.h[0] = *(const v8h*)(wq + (size_t)nc * K + k0 + kh * 8);
    bf.h[1] = *(const v8h*)(wq + (size_t)nc * K + k0 + 16 + kh * 8);
    acc = __builtin_amdgcn_wmma_f32_16x16x32_f16(
        false, af.v, false, bf.v, (short)0, acc, false, false);
  }
#pragma unroll
  for (int r = 0; r < 8; ++r) {
    int m = r + 8 * kh;
    if (n < N) out[m * N + n] = acc[r] + bias[n];
  }
}

// ---------------------------------------------------------------------------
// Host orchestration (all activations NHWC)
// ---------------------------------------------------------------------------
extern "C" void kernel_launch(void* const* d_in, const int* in_sizes, int n_in,
                              void* d_out, int out_size, void* d_ws, size_t ws_size,
                              hipStream_t stream) {
  (void)in_sizes; (void)n_in; (void)out_size; (void)ws_size;

  static const int BCFG[4] = {6, 12, 24, 16};
  const int GROWTH = 32, BOT = 128;

  // -------- workspace partition --------
  char* ws = (char*)d_ws;
  _Float16* FEATS  = (_Float16*)(ws);                 // 28 MB concat buffer
  _Float16* QBUF   = (_Float16*)(ws + (28u << 20));   // 28 MB quantized acts
  _Float16* TMP    = (_Float16*)(ws + (56u << 20));   // 28 MB raw conv outs
  _Float16* WQ     = (_Float16*)(ws + (84u << 20));   // 4 MB K-padded weights
  unsigned* MAXAB  = (unsigned*)(ws + (88u << 20));
  float*    POOLED = (float*)(ws + (88u << 20) + 256);

  int pi = 0;
  auto P = [&]() -> const float* { return (const float*)d_in[pi++]; };

  auto quant_weight = [&](const float* w, int Cout, int C, int KH, int KW) -> int {
    int Ktot = C * KH * KW;
    int Kpad = (Ktot + 31) & ~31;
    int n = Cout * Ktot;
    zero_u32_kernel<<<1, 1, 0, stream>>>(MAXAB);
    int blk = (n + 255) / 256; if (blk > 1024) blk = 1024;
    maxabs_kernel<<<blk, 256, 0, stream>>>(w, n, MAXAB);
    int total = Cout * Kpad;
    quant_weight_kernel<<<(total + 255) / 256, 256, 0, stream>>>(
        w, MAXAB, WQ, Cout, C, KH, KW, Kpad);
    return Kpad;
  };

  auto bnq = [&](const _Float16* in, int C, int NPX, int inCS,
                 const float* g, const float* b, const float* m, const float* v,
                 _Float16* out) {
    int total = NPX * (C >> 3);
    bn_relu_q_kernel<<<(total + 255) / 256, 256, 0, stream>>>(
        in, g, b, m, v, out, C, NPX, inCS);
  };

  auto conv1x1 = [&](const _Float16* in, _Float16* out, int Cin, int Mtot,
                     int Cout, int CS, int cofs, int Kpad) {
    dim3 g((Mtot + CONV_BM - 1) / CONV_BM, (Cout + 63) / 64);
    conv1x1_nhwc<<<g, 128, 0, stream>>>(in, WQ, out, Cin, Mtot, Cout, CS, cofs, Kpad);
  };

  // -------- stem --------
  const float* x = P();
  const float* w_first = P();
  quant_input_kernel<<<(16 * 224 * 224 * 3 + 255) / 256, 256, 0, stream>>>(
      x, QBUF, 224 * 224);
  {
    int Kpad = quant_weight(w_first, 64, 3, 7, 7);    // Kpad = 160
    int M = 16 * 112 * 112;
    dim3 g(M / CONV_BM, 1);
    conv7x7s2_nhwc<<<g, 128, 0, stream>>>(QBUF, WQ, TMP, Kpad);
  }
  {
    int total = 16 * 56 * 56 * (64 >> 3);
    maxpool3s2p1_kernel<<<(total + 255) / 256, 256, 0, stream>>>(
        TMP, FEATS, 64, 112, 112, 56, 56, /*CS=*/256);
  }

  int H = 56, W = 56, nf = 64, CS = 256;

  // -------- dense blocks --------
  for (int bi = 0; bi < 4; ++bi) {
    int nl = BCFG[bi];
    for (int li = 0; li < nl; ++li) {
      int cin = nf + li * GROWTH;
      const float* g1 = P(); const float* b1 = P();
      const float* m1 = P(); const float* v1 = P();
      const float* c1w = P();
      const float* g2 = P(); const float* b2 = P();
      const float* m2 = P(); const float* v2 = P();
      const float* c2w = P();

      int NPX = 16 * H * W;
      // bottleneck 1x1: cin -> 128
      bnq(FEATS, cin, NPX, CS, g1, b1, m1, v1, QBUF);
      {
        int Kpad = quant_weight(c1w, BOT, cin, 1, 1);
        conv1x1(QBUF, TMP, cin, NPX, BOT, BOT, 0, Kpad);
      }
      // 3x3: 128 -> 32, written into concat slice
      bnq(TMP, BOT, NPX, BOT, g2, b2, m2, v2, QBUF);
      {
        int Kpad = quant_weight(c2w, GROWTH, BOT, 3, 3);
        dim3 g((NPX + CONV_BM - 1) / CONV_BM, 1);
        conv3x3_c128_nhwc<<<g, 128, 0, stream>>>(
            QBUF, WQ, FEATS, H, W, GROWTH, CS, cin, Kpad);
      }
    }
    nf += nl * GROWTH;
    if (bi != 3) {
      const float* tg = P(); const float* tb = P();
      const float* tm = P(); const float* tv = P();
      const float* tw = P();
      int NPX = 16 * H * W;
      bnq(FEATS, nf, NPX, CS, tg, tb, tm, tv, QBUF);
      {
        int Kpad = quant_weight(tw, nf / 2, nf, 1, 1);
        conv1x1(QBUF, TMP, nf, NPX, nf / 2, nf / 2, 0, Kpad);
      }
      int CSn = nf / 2 + BCFG[bi + 1] * GROWTH;
      int total = 16 * (H / 2) * (W / 2) * ((nf / 2) >> 3);
      avgpool2_kernel<<<(total + 255) / 256, 256, 0, stream>>>(
          TMP, FEATS, nf / 2, H, W, CSn);
      H >>= 1; W >>= 1; nf >>= 1; CS = CSn;
    }
  }

  // -------- head --------
  const float* lg = P(); const float* lb = P();
  const float* lm = P(); const float* lv = P();
  const float* fc_w = P();
  const float* fc_b = P();

  bnq(FEATS, nf, 16 * H * W, CS, lg, lb, lm, lv, QBUF);   // nf = 1024 @ 7x7
  {
    int total = 16 * (nf >> 3);
    gpool_kernel<<<(total + 255) / 256, 256, 0, stream>>>(QBUF, POOLED, nf, H * W);
  }
  {
    (void)quant_weight(fc_w, 1000, nf, 1, 1);             // Kpad == 1024
    fc_wmma_kernel<<<(1000 + 15) / 16, 32, 0, stream>>>(
        POOLED, WQ, fc_b, (float*)d_out, nf, 1000);
  }
}
